// OptimizedTritonKroneckerLinear_48395691491920
// MI455X (gfx1250) — matrix operand order repaired and checked by reading, not compiled, not run on hardware
//
#include <hip/hip_runtime.h>

// Y = X @ (A kron B)^T + bias, fused two-stage contraction, bf16x3 (split-fp32)
// WMMA on gfx1250. One wave32 per workgroup; each wave processes ROWS_PER_BLOCK rows.

#define MDIM 8192
#define KDIM 4096
#define NDIM 4096
#define LDK  72      // LDS row stride in bf16 elems: 144 B = 9*16 -> 16B aligned rows
#define RPB  8       // rows per block

typedef __attribute__((ext_vector_type(16))) __bf16 v16bf;
typedef __attribute__((ext_vector_type(8)))  __bf16 v8bf;
typedef __attribute__((ext_vector_type(4)))  __bf16 v4bf;
typedef __attribute__((ext_vector_type(8)))  float  v8f;
typedef __attribute__((ext_vector_type(4)))  float  v4f;

union V16U { v16bf v; v8bf h[2]; };

// ---- fragment loaders (wave32 WMMA 16x16x32 bf16 layouts) ----
// A-operand (16 x 32 tile of a row-major [M][K] matrix, row stride ld):
//   lane(half,l): elems 0..7  = S[row0+l][k0 + 8*half + 0..7]
//                 elems 8..15 = S[row0+l][k0 + 16 + 8*half + 0..7]
__device__ __forceinline__ v16bf frag_a(const __bf16* S, int ld, int row0, int k0,
                                        int l, int half) {
  const __bf16* p = S + (row0 + l) * ld + k0 + 8 * half;
  V16U u;
  u.h[0] = *(const v8bf*)(p);
  u.h[1] = *(const v8bf*)(p + 16);
  return u.v;
}

// B-operand (32 x 16 tile) read from TRANSPOSED storage St (row-major [N][K]):
//   lane(half,l): elems 0..7  = St[col0+l][k0 + 8*half + 0..7]
//                 elems 8..15 = St[col0+l][k0 + 16 + 8*half + 0..7]
__device__ __forceinline__ v16bf frag_b(const __bf16* St, int ld, int k0, int col0,
                                        int l, int half) {
  const __bf16* p = St + (col0 + l) * ld + k0 + 8 * half;
  V16U u;
  u.h[0] = *(const v8bf*)(p);
  u.h[1] = *(const v8bf*)(p + 16);
  return u.v;
}

__device__ __forceinline__ v8f wmma_bf16(v16bf a, v16bf b, v8f c) {
  return __builtin_amdgcn_wmma_f32_16x16x32_bf16(
      /*neg_a=*/false, a, /*neg_b=*/false, b,
      /*c_mod=*/(short)0, c, /*reuse_a=*/false, /*reuse_b=*/false);
}

// bf16x3: (ah+al)(bh+bl) ~= ah*bh + ah*bl + al*bh, fp32 accumulate
__device__ __forceinline__ v8f mma3(v16bf ah, v16bf al, v16bf bh, v16bf bl, v8f c) {
  c = wmma_bf16(al, bh, c);
  c = wmma_bf16(ah, bl, c);
  c = wmma_bf16(ah, bh, c);
  return c;
}

// ---- pre-kernel: split A and B (fp32) into hi/lo bf16 planes in workspace ----
// ws layout (bf16): [0..4095]=Ah [4096..]=Al [8192..]=Bh [12288..]=Bl
__global__ void __launch_bounds__(256) split_weights_kernel(
    const float* __restrict__ A, const float* __restrict__ B,
    __bf16* __restrict__ ws) {
  const int i = blockIdx.x * blockDim.x + threadIdx.x;
  if (i < 4096) {
    const float a  = A[i];
    const __bf16 ah = (__bf16)a;
    ws[i]          = ah;
    ws[4096 + i]   = (__bf16)(a - (float)ah);
    const float b  = B[i];
    const __bf16 bh = (__bf16)b;
    ws[8192 + i]   = bh;
    ws[12288 + i]  = (__bf16)(b - (float)bh);
  }
}

// ---- fused main kernel: one wave per block ----
__global__ void __launch_bounds__(32) kron_fused_kernel(
    const float* __restrict__ X,
    const __bf16* __restrict__ W,     // split weights from d_ws
    const float* __restrict__ bias,
    float* __restrict__ Y) {
  __shared__ __bf16 sXh[64 * LDK];   // X row,  hi plane, row-major [a][b]
  __shared__ __bf16 sXl[64 * LDK];   // X row,  lo plane
  __shared__ __bf16 sTh[64 * LDK];   // T^T,    hi plane, row-major [q][a]
  __shared__ __bf16 sTl[64 * LDK];   // T^T,    lo plane

  const __bf16* Ah = W;
  const __bf16* Al = W + 4096;
  const __bf16* Bh = W + 8192;
  const __bf16* Bl = W + 12288;

  const int lane = (int)(threadIdx.x & 31u);
  const int half = lane >> 4;
  const int l    = lane & 15;

  for (int rb = 0; rb < RPB; ++rb) {
    const int m = blockIdx.x * RPB + rb;
    const float* xrow = X + (size_t)m * KDIM;

    // ---- stage X row (16 KB) into LDS with hi/lo split; coalesced b128 loads ----
    #pragma unroll
    for (int it = 0; it < 32; ++it) {
      const int flat = (it * 32 + lane) * 4;       // consecutive lanes -> 16B apart
      const v4f x = *(const v4f*)(xrow + flat);
      const int r = flat >> 6, c = flat & 63;
      v4bf hi, lo;
      #pragma unroll
      for (int k = 0; k < 4; ++k) {
        const float  xv = x[k];
        const __bf16 h  = (__bf16)xv;
        hi[k] = h;
        lo[k] = (__bf16)(xv - (float)h);
      }
      *(v4bf*)(sXh + r * LDK + c) = hi;
      *(v4bf*)(sXl + r * LDK + c) = lo;
    }
    // prefetch next row's X into L2 while we compute this one
    if (rb + 1 < RPB) {
      const float* nxt = xrow + KDIM;
      #pragma unroll
      for (int pf = 0; pf < 4; ++pf)
        __builtin_prefetch(nxt + (pf * 32 + lane) * 32, 0, 0);  // 128 x 128B lines
    }
    __syncthreads();

    // ---- GEMM1': T^T = B * X^T  (D rows = q, cols = a); bf16x3, fp32 accum ----
    for (int qi = 0; qi < 4; ++qi) {
      v8f c0 = {}, c1 = {}, c2 = {}, c3 = {};
      #pragma unroll
      for (int ks = 0; ks < 2; ++ks) {
        const int k0 = ks * 32;                               // K = b
        const v16bf bh = frag_a(Bh, 64, qi * 16, k0, l, half);
        const v16bf bl = frag_a(Bl, 64, qi * 16, k0, l, half);
        const v16bf x0h = frag_b(sXh, LDK, k0, 0,  l, half);
        const v16bf x0l = frag_b(sXl, LDK, k0, 0,  l, half);
        c0 = mma3(bh, bl, x0h, x0l, c0);
        const v16bf x1h = frag_b(sXh, LDK, k0, 16, l, half);
        const v16bf x1l = frag_b(sXl, LDK, k0, 16, l, half);
        c1 = mma3(bh, bl, x1h, x1l, c1);
        const v16bf x2h = frag_b(sXh, LDK, k0, 32, l, half);
        const v16bf x2l = frag_b(sXl, LDK, k0, 32, l, half);
        c2 = mma3(bh, bl, x2h, x2l, c2);
        const v16bf x3h = frag_b(sXh, LDK, k0, 48, l, half);
        const v16bf x3l = frag_b(sXl, LDK, k0, 48, l, half);
        c3 = mma3(bh, bl, x3h, x3l, c3);
      }
      // split+store T^T band (rows q = qi*16 .. +15), C layout: M = r + 8*half
      #pragma unroll
      for (int r = 0; r < 8; ++r) {
        const int row = qi * 16 + half * 8 + r;
        __bf16* ph = sTh + row * LDK + l;
        __bf16* pl = sTl + row * LDK + l;
        { const float v = c0[r]; const __bf16 h = (__bf16)v; ph[0]  = h; pl[0]  = (__bf16)(v - (float)h); }
        { const float v = c1[r]; const __bf16 h = (__bf16)v; ph[16] = h; pl[16] = (__bf16)(v - (float)h); }
        { const float v = c2[r]; const __bf16 h = (__bf16)v; ph[32] = h; pl[32] = (__bf16)(v - (float)h); }
        { const float v = c3[r]; const __bf16 h = (__bf16)v; ph[48] = h; pl[48] = (__bf16)(v - (float)h); }
      }
    }
    __syncthreads();

    // ---- GEMM2: Y_m = A * T  (K = a), T read from T^T storage; bf16x3 ----
    float* yrow = Y + (size_t)m * NDIM;
    for (int pi = 0; pi < 4; ++pi) {
      v8f d0 = {}, d1 = {}, d2 = {}, d3 = {};
      #pragma unroll
      for (int ks = 0; ks < 2; ++ks) {
        const int k0 = ks * 32;                               // K = a
        const v16bf ah = frag_a(Ah, 64, pi * 16, k0, l, half);
        const v16bf al = frag_a(Al, 64, pi * 16, k0, l, half);
        const v16bf t0h = frag_b(sTh, LDK, k0, 0,  l, half);
        const v16bf t0l = frag_b(sTl, LDK, k0, 0,  l, half);
        d0 = mma3(ah, al, t0h, t0l, d0);
        const v16bf t1h = frag_b(sTh, LDK, k0, 16, l, half);
        const v16bf t1l = frag_b(sTl, LDK, k0, 16, l, half);
        d1 = mma3(ah, al, t1h, t1l, d1);
        const v16bf t2h = frag_b(sTh, LDK, k0, 32, l, half);
        const v16bf t2l = frag_b(sTl, LDK, k0, 32, l, half);
        d2 = mma3(ah, al, t2h, t2l, d2);
        const v16bf t3h = frag_b(sTh, LDK, k0, 48, l, half);
        const v16bf t3l = frag_b(sTl, LDK, k0, 48, l, half);
        d3 = mma3(ah, al, t3h, t3l, d3);
      }
      // store Y tiles with bias; coalesced 64B half-wave segments
      #pragma unroll
      for (int r = 0; r < 8; ++r) {
        const int p    = pi * 16 + half * 8 + r;
        const int base = p * 64 + l;                          // n = q
        yrow[base]      = d0[r] + bias[base];
        yrow[base + 16] = d1[r] + bias[base + 16];
        yrow[base + 32] = d2[r] + bias[base + 32];
        yrow[base + 48] = d3[r] + bias[base + 48];
      }
    }
    __syncthreads();
  }
}

extern "C" void kernel_launch(void* const* d_in, const int* in_sizes, int n_in,
                              void* d_out, int out_size, void* d_ws, size_t ws_size,
                              hipStream_t stream) {
  (void)in_sizes; (void)n_in; (void)out_size; (void)ws_size;
  const float* x    = (const float*)d_in[0];
  const float* A    = (const float*)d_in[1];
  const float* B    = (const float*)d_in[2];
  const float* bias = (const float*)d_in[3];
  float*  y  = (float*)d_out;
  __bf16* ws = (__bf16*)d_ws;   // 32 KB used: Ah|Al|Bh|Bl

  split_weights_kernel<<<16, 256, 0, stream>>>(A, B, ws);
  kron_fused_kernel<<<MDIM / RPB, 32, 0, stream>>>(x, ws, bias, y);
}
